// DecoderRNN_63402307224246
// MI455X (gfx1250) — matrix operand order repaired and checked by reading, not compiled, not run on hardware
//
#include <hip/hip_runtime.h>
#include <hip/hip_bf16.h>

// ---------------------------------------------------------------------------
// DecoderRNN (attention LSTM) for MI455X / gfx1250.
// All GEMMs use v_wmma_f32_16x16x32_bf16 (fp32 accumulate). Weights/activations
// converted to bf16 once (weights) / per step (activations). GEMM blocks MT
// M-tiles per wave so each weight (B) fragment is loaded once and used by MT
// WMMAs: MT=2 for the M=32 recurrence GEMMs (weights read exactly once per
// step, L2-resident), MT=4 for the large-M GEMMs (a1 precompute, final FC) to
// cut the redundant L2 weight stream (Wfc: 20 -> 10 full reads).
// Shapes: B=32 N=196 ENC=2048 E=512 H=1024 A=256 V=32000 T=20 L=2
// ---------------------------------------------------------------------------

#define Bz   32
#define Nn   196
#define ENCz 2048
#define Ez   512
#define Hz   1024
#define Az   256
#define Vz   32000
#define Tz   20

typedef __bf16 bf16_t;
typedef __attribute__((ext_vector_type(16))) __bf16 v16bf;
typedef __attribute__((ext_vector_type(8)))  __bf16 v8bf;
typedef __attribute__((ext_vector_type(8)))  float  v8f;

__device__ __forceinline__ bf16_t f2bf(float f) {
    unsigned u = __float_as_uint(f);
    unsigned r = (u + 0x7FFFu + ((u >> 16) & 1u)) >> 16;
    unsigned short s = (unsigned short)r;
    return __builtin_bit_cast(bf16_t, s);
}
__device__ __forceinline__ float sigmoidf_(float x) {
    return 1.0f / (1.0f + __expf(-x));
}

// ---------------------------------------------------------------------------
// bf16 WMMA GEMM:  C[M,N] = A[M,K] * W[N,K]^T (+ bias[N])
// One wave computes an (MT*16)x16 strip: one B fragment feeds MT WMMAs.
// 4 waves per block take 4 adjacent N tiles.
// M % (16*MT) == 0, N % 64 == 0, K % 32 == 0 for all uses here.
// ---------------------------------------------------------------------------
template <int MT>
__global__ __launch_bounds__(128)
void k_gemm_bf16(const bf16_t* __restrict__ A, const bf16_t* __restrict__ W,
                 const float* __restrict__ bias, float* __restrict__ C,
                 int M, int N, int K)
{
    const int lane = threadIdx.x & 31;
    const int wave = threadIdx.x >> 5;
    const int tn   = blockIdx.x * 4 + wave;   // N tile index
    const int tm   = blockIdx.y;              // (MT*16)-row M strip index
    const int r    = lane & 15;
    const int hi   = lane >> 4;

    const bf16_t* Arow[MT];
#pragma unroll
    for (int m = 0; m < MT; ++m)
        Arow[m] = A + (size_t)(tm * (MT * 16) + m * 16 + r) * K;
    const bf16_t* Brow = W + (size_t)(tn * 16 + r) * K + hi * 16;

    v8f acc[MT];
#pragma unroll
    for (int m = 0; m < MT; ++m) acc[m] = (v8f){};

    for (int k0 = 0; k0 < K; k0 += 32) {
        // B fragment (32x16): lane column n=r, K = hi*16 + 0..15 contiguous in W row
        const v16bf b = *(const v16bf*)(Brow + k0);
#pragma unroll
        for (int m = 0; m < MT; ++m) {
            // A fragment (16x32): lane row m=r, K = hi*16 + 0..15 contiguous (matches B)
            const v8bf a0 = *(const v8bf*)(Arow[m] + k0 + hi * 16);
            const v8bf a1 = *(const v8bf*)(Arow[m] + k0 + hi * 16 + 8);
            v16bf a;
#pragma unroll
            for (int i = 0; i < 8; ++i) { a[i] = a0[i]; a[8 + i] = a1[i]; }
            acc[m] = __builtin_amdgcn_wmma_f32_16x16x32_bf16(
                         false, a, false, b, (short)0, acc[m], false, false);
        }
    }
    const float bv = bias ? bias[tn * 16 + r] : 0.0f;
#pragma unroll
    for (int m = 0; m < MT; ++m) {
        float* Crow = C + (size_t)(tm * (MT * 16) + m * 16 + hi * 8) * N
                        + (size_t)tn * 16 + r;
#pragma unroll
        for (int v = 0; v < 8; ++v) Crow[(size_t)v * N] = acc[m][v] + bv;
    }
}

// ---------------------------------------------------------------------------
// Setup / conversion kernels
// ---------------------------------------------------------------------------
__global__ void k_cvt_bf16(const float* __restrict__ src, bf16_t* __restrict__ dst, long n)
{
    long i = (long)blockIdx.x * blockDim.x + threadIdx.x;
    if (i < n) dst[i] = f2bf(src[i]);
}

// dst[n, 0:K1]=W1[n,:], dst[n, K1:K1+K2]=W2[n,:]  (concat LSTM Wih|Whh along K)
__global__ void k_concat_cvt(const float* __restrict__ W1, int K1,
                             const float* __restrict__ W2, int K2,
                             bf16_t* __restrict__ dst, long total)
{
    long i = (long)blockIdx.x * blockDim.x + threadIdx.x;
    if (i >= total) return;
    int K = K1 + K2;
    long n = i / K;
    int  k = (int)(i - n * K);
    float v = (k < K1) ? W1[n * (long)K1 + k] : W2[n * (long)K2 + (k - K1)];
    dst[i] = f2bf(v);
}

__global__ void k_add2(const float* __restrict__ a, const float* __restrict__ b,
                       float* __restrict__ o, int n)
{
    int i = blockIdx.x * blockDim.x + threadIdx.x;
    if (i < n) o[i] = a[i] + b[i];
}

// mean over N and bf16 copy of it (A operand for the init GEMMs)
__global__ __launch_bounds__(256)
void k_mean_cvt(const float* __restrict__ feats, float* __restrict__ mean,
                bf16_t* __restrict__ mean_bf)
{
    int idx = blockIdx.x * blockDim.x + threadIdx.x;      // B*ENC = 65536
    int b = idx >> 11, d = idx & 2047;
    const float* p = feats + (size_t)b * Nn * ENCz + d;
    float s = 0.f;
    for (int n = 0; n < Nn; ++n) s += p[(size_t)n * ENCz];
    s *= (1.0f / (float)Nn);
    mean[idx] = s;
    mean_bf[idx] = f2bf(s);
}

// reshape(L,B,H) is a flat reinterpret of [B, L*H]:
//   h0 = Yh_flat[0 : B*H], h1 = Yh_flat[B*H : 2*B*H]   (same for c)
__global__ __launch_bounds__(256)
void k_init_state(const float* __restrict__ Yh, const float* __restrict__ Yc,
                  float* __restrict__ c0, float* __restrict__ c1,
                  bf16_t* __restrict__ h0_bf, bf16_t* __restrict__ x_bf,
                  bf16_t* __restrict__ x1_bf)
{
    int idx = blockIdx.x * blockDim.x + threadIdx.x;      // B*H = 32768
    int b = idx >> 10, h = idx & 1023;
    c0[idx] = Yc[idx];
    c1[idx] = Yc[Bz * Hz + idx];
    bf16_t h0b = f2bf(Yh[idx]);
    h0_bf[idx] = h0b;
    x_bf[(size_t)b * (Ez + ENCz + Hz) + Ez + ENCz + h] = h0b;
    x1_bf[(size_t)b * (2 * Hz) + h]        = h0b;
    x1_bf[(size_t)b * (2 * Hz) + Hz + h]   = f2bf(Yh[Bz * Hz + idx]);
}

// ---------------------------------------------------------------------------
// Per-step kernels
// ---------------------------------------------------------------------------
// Bahdanau attention: scores = va . tanh(a1[b,n,:] + a2[b,:]) + bv ; softmax;
// ctx[b,:] = sum_n w[n] * features[b,n,:]
__global__ __launch_bounds__(256)
void k_attention(const float* __restrict__ a1, const float* __restrict__ a2,
                 const float* __restrict__ va, const float* __restrict__ bvp,
                 const float* __restrict__ features, float* __restrict__ ctx)
{
    __shared__ float s_a2[Az];
    __shared__ float s_va[Az];
    __shared__ float s_w[Nn];
    __shared__ float s_red[256];
    const int b = blockIdx.x, tid = threadIdx.x;
    s_a2[tid] = a2[b * Az + tid];
    s_va[tid] = va[tid];
    __syncthreads();

    float score = -3.4e38f;
    if (tid < Nn) {
        const float* row = a1 + ((size_t)b * Nn + tid) * Az;
        float s = 0.f;
        for (int a = 0; a < Az; ++a) s += s_va[a] * tanhf(row[a] + s_a2[a]);
        score = s + bvp[0];
    }
    s_red[tid] = score; __syncthreads();
    for (int off = 128; off > 0; off >>= 1) {
        if (tid < off) s_red[tid] = fmaxf(s_red[tid], s_red[tid + off]);
        __syncthreads();
    }
    const float mx = s_red[0]; __syncthreads();
    const float e = (tid < Nn) ? __expf(score - mx) : 0.f;
    s_red[tid] = e; __syncthreads();
    for (int off = 128; off > 0; off >>= 1) {
        if (tid < off) s_red[tid] += s_red[tid + off];
        __syncthreads();
    }
    const float inv = 1.0f / s_red[0];
    if (tid < Nn) s_w[tid] = e * inv;
    __syncthreads();

    const float* fb = features + (size_t)b * Nn * ENCz;
    for (int d = tid; d < ENCz; d += 256) {
        float acc = 0.f;
        for (int n = 0; n < Nn; ++n) acc += s_w[n] * fb[(size_t)n * ENCz + d];
        ctx[b * ENCz + d] = acc;
    }
}

// x = [ embed(caption_t) | ctx ] as bf16 (h0 slice of x_bf is written by LSTM0)
__global__ void k_build_x(const float* __restrict__ emb, const int* __restrict__ captions,
                          const float* __restrict__ ctx, bf16_t* __restrict__ x_bf, int t)
{
    int idx = blockIdx.x * blockDim.x + threadIdx.x;      // B*(E+ENC) = 81920
    if (idx >= Bz * (Ez + ENCz)) return;
    int b = idx / (Ez + ENCz), j = idx % (Ez + ENCz);
    float v;
    if (j < Ez) {
        int tok = captions[b * Tz + t];
        v = emb[(size_t)tok * Ez + j];
    } else {
        v = ctx[b * ENCz + (j - Ez)];
    }
    x_bf[(size_t)b * (Ez + ENCz + Hz) + j] = f2bf(v);
}

// LSTM layer 0 pointwise: gates order i,f,g,o (PyTorch)
__global__ __launch_bounds__(256)
void k_lstm_point0(const float* __restrict__ gates, float* __restrict__ c0,
                   bf16_t* __restrict__ h0_bf, bf16_t* __restrict__ x_bf,
                   bf16_t* __restrict__ x1_bf)
{
    int idx = blockIdx.x * blockDim.x + threadIdx.x;      // B*H
    int b = idx >> 10, h = idx & 1023;
    const float* g = gates + (size_t)b * 4 * Hz;
    float gi = sigmoidf_(g[h]);
    float gf = sigmoidf_(g[Hz + h]);
    float gg = tanhf(g[2 * Hz + h]);
    float go = sigmoidf_(g[3 * Hz + h]);
    float c  = gf * c0[idx] + gi * gg;
    c0[idx] = c;
    float hh = go * tanhf(c);
    bf16_t hb = f2bf(hh);
    h0_bf[idx] = hb;
    x_bf[(size_t)b * (Ez + ENCz + Hz) + Ez + ENCz + h] = hb;   // for next step's gates0
    x1_bf[(size_t)b * (2 * Hz) + h] = hb;                      // layer-1 input
}

// LSTM layer 1 pointwise; h1 feeds x1 (next step) and the output sequence
__global__ __launch_bounds__(256)
void k_lstm_point1(const float* __restrict__ gates, float* __restrict__ c1,
                   bf16_t* __restrict__ x1_bf, bf16_t* __restrict__ outs_bf, int t)
{
    int idx = blockIdx.x * blockDim.x + threadIdx.x;      // B*H
    int b = idx >> 10, h = idx & 1023;
    const float* g = gates + (size_t)b * 4 * Hz;
    float gi = sigmoidf_(g[h]);
    float gf = sigmoidf_(g[Hz + h]);
    float gg = tanhf(g[2 * Hz + h]);
    float go = sigmoidf_(g[3 * Hz + h]);
    float c  = gf * c1[idx] + gi * gg;
    c1[idx] = c;
    float hh = go * tanhf(c);
    bf16_t hb = f2bf(hh);
    x1_bf[(size_t)b * (2 * Hz) + Hz + h] = hb;
    outs_bf[((size_t)b * Tz + t) * Hz + h] = hb;
}

// ---------------------------------------------------------------------------
// Host launcher
// ---------------------------------------------------------------------------
extern "C" void kernel_launch(void* const* d_in, const int* in_sizes, int n_in,
                              void* d_out, int out_size, void* d_ws, size_t ws_size,
                              hipStream_t stream)
{
    const float* features = (const float*)d_in[0];
    const int*   captions = (const int*)d_in[1];
    const float* emb      = (const float*)d_in[2];
    const float* Wa       = (const float*)d_in[3];
    const float* ba       = (const float*)d_in[4];
    const float* Ua       = (const float*)d_in[5];
    const float* bu       = (const float*)d_in[6];
    const float* va       = (const float*)d_in[7];
    const float* bv       = (const float*)d_in[8];
    const float* Wh_init  = (const float*)d_in[9];
    const float* bh_init  = (const float*)d_in[10];
    const float* Wc_init  = (const float*)d_in[11];
    const float* bc_init  = (const float*)d_in[12];
    const float* Wih0     = (const float*)d_in[13];
    const float* Whh0     = (const float*)d_in[14];
    const float* bih0     = (const float*)d_in[15];
    const float* bhh0     = (const float*)d_in[16];
    const float* Wih1     = (const float*)d_in[17];
    const float* Whh1     = (const float*)d_in[18];
    const float* bih1     = (const float*)d_in[19];
    const float* bhh1     = (const float*)d_in[20];
    const float* Wfc      = (const float*)d_in[21];
    const float* bfc      = (const float*)d_in[22];
    float* out = (float*)d_out;   // [B, T, V]

    // ---- workspace partition (~170 MB) ----
    size_t off = 0;
    char* base = (char*)d_ws;
    auto alloc = [&](size_t bytes) -> char* {
        char* p = base + off;
        off += (bytes + 255) & ~(size_t)255;
        return p;
    };
    const int KX0 = Ez + ENCz + Hz;     // 3584, K of gates0 GEMM
    const int KX1 = 2 * Hz;             // 2048, K of gates1 GEMM

    bf16_t* feat_bf = (bf16_t*)alloc((size_t)Bz * Nn * ENCz * 2);
    bf16_t* Wa_bf   = (bf16_t*)alloc((size_t)Az * ENCz * 2);
    bf16_t* Ua_bf   = (bf16_t*)alloc((size_t)Az * Hz * 2);
    bf16_t* Wh_bf   = (bf16_t*)alloc((size_t)2 * Hz * ENCz * 2);
    bf16_t* Wc_bf   = (bf16_t*)alloc((size_t)2 * Hz * ENCz * 2);
    bf16_t* Wcat0_bf= (bf16_t*)alloc((size_t)4 * Hz * KX0 * 2);
    bf16_t* Wcat1_bf= (bf16_t*)alloc((size_t)4 * Hz * KX1 * 2);
    bf16_t* Wfc_bf  = (bf16_t*)alloc((size_t)Vz * Hz * 2);
    float*  meanf   = (float*)alloc((size_t)Bz * ENCz * 4);
    bf16_t* mean_bf = (bf16_t*)alloc((size_t)Bz * ENCz * 2);
    float*  Yh      = (float*)alloc((size_t)Bz * 2 * Hz * 4);
    float*  Yc      = (float*)alloc((size_t)Bz * 2 * Hz * 4);
    float*  a1      = (float*)alloc((size_t)Bz * Nn * Az * 4);
    float*  a2      = (float*)alloc((size_t)Bz * Az * 4);
    float*  ctx     = (float*)alloc((size_t)Bz * ENCz * 4);
    bf16_t* x_bf    = (bf16_t*)alloc((size_t)Bz * KX0 * 2);
    bf16_t* x1_bf   = (bf16_t*)alloc((size_t)Bz * KX1 * 2);
    bf16_t* h0_bf   = (bf16_t*)alloc((size_t)Bz * Hz * 2);
    float*  gates0  = (float*)alloc((size_t)Bz * 4 * Hz * 4);
    float*  gates1  = (float*)alloc((size_t)Bz * 4 * Hz * 4);
    float*  c0      = (float*)alloc((size_t)Bz * Hz * 4);
    float*  c1      = (float*)alloc((size_t)Bz * Hz * 4);
    bf16_t* outs_bf = (bf16_t*)alloc((size_t)Bz * Tz * Hz * 2);
    float*  bias0   = (float*)alloc((size_t)4 * Hz * 4);
    float*  bias1   = (float*)alloc((size_t)4 * Hz * 4);
    (void)ws_size; (void)n_in; (void)in_sizes; (void)out_size;

    const int TB = 256;
    auto blk = [](long n, int t) { return (unsigned)((n + t - 1) / t); };

    // ---- one-time conversions (cheap vs. 20-step recurrence) ----
    k_mean_cvt<<<blk(Bz * ENCz, TB), TB, 0, stream>>>(features, meanf, mean_bf);
    k_cvt_bf16<<<blk((long)Bz * Nn * ENCz, TB), TB, 0, stream>>>(features, feat_bf, (long)Bz * Nn * ENCz);
    k_cvt_bf16<<<blk((long)Az * ENCz, TB), TB, 0, stream>>>(Wa, Wa_bf, (long)Az * ENCz);
    k_cvt_bf16<<<blk((long)Az * Hz, TB), TB, 0, stream>>>(Ua, Ua_bf, (long)Az * Hz);
    k_cvt_bf16<<<blk((long)2 * Hz * ENCz, TB), TB, 0, stream>>>(Wh_init, Wh_bf, (long)2 * Hz * ENCz);
    k_cvt_bf16<<<blk((long)2 * Hz * ENCz, TB), TB, 0, stream>>>(Wc_init, Wc_bf, (long)2 * Hz * ENCz);
    k_cvt_bf16<<<blk((long)Vz * Hz, TB), TB, 0, stream>>>(Wfc, Wfc_bf, (long)Vz * Hz);
    k_concat_cvt<<<blk((long)4 * Hz * KX0, TB), TB, 0, stream>>>(Wih0, Ez + ENCz, Whh0, Hz, Wcat0_bf, (long)4 * Hz * KX0);
    k_concat_cvt<<<blk((long)4 * Hz * KX1, TB), TB, 0, stream>>>(Wih1, Hz, Whh1, Hz, Wcat1_bf, (long)4 * Hz * KX1);
    k_add2<<<blk(4 * Hz, TB), TB, 0, stream>>>(bih0, bhh0, bias0, 4 * Hz);
    k_add2<<<blk(4 * Hz, TB), TB, 0, stream>>>(bih1, bhh1, bias1, 4 * Hz);

    // ---- a1 = features @ Wa^T + ba : M=6272, N=256, K=2048 (MT=4) ----
    k_gemm_bf16<4><<<dim3(Az / 64, (Bz * Nn) / 64), 128, 0, stream>>>(
        feat_bf, Wa_bf, ba, a1, Bz * Nn, Az, ENCz);

    // ---- init hidden/cell: Y = mean @ W^T + b : M=32, N=2048, K=2048 ----
    k_gemm_bf16<2><<<dim3((2 * Hz) / 64, 1), 128, 0, stream>>>(
        mean_bf, Wh_bf, bh_init, Yh, Bz, 2 * Hz, ENCz);
    k_gemm_bf16<2><<<dim3((2 * Hz) / 64, 1), 128, 0, stream>>>(
        mean_bf, Wc_bf, bc_init, Yc, Bz, 2 * Hz, ENCz);
    k_init_state<<<blk(Bz * Hz, TB), TB, 0, stream>>>(Yh, Yc, c0, c1, h0_bf, x_bf, x1_bf);

    // ---- recurrence ----
    for (int t = 0; t < Tz; ++t) {
        // a2 = h0 @ Ua^T + bu : M=32, N=256, K=1024
        k_gemm_bf16<2><<<dim3(Az / 64, 1), 128, 0, stream>>>(
            h0_bf, Ua_bf, bu, a2, Bz, Az, Hz);
        k_attention<<<Bz, 256, 0, stream>>>(a1, a2, va, bv, features, ctx);
        k_build_x<<<blk((long)Bz * (Ez + ENCz), TB), TB, 0, stream>>>(emb, captions, ctx, x_bf, t);
        // gates0 = [e|ctx|h0] @ [Wih0|Whh0]^T + (bih0+bhh0) : M=32, N=4096, K=3584
        k_gemm_bf16<2><<<dim3((4 * Hz) / 64, 1), 128, 0, stream>>>(
            x_bf, Wcat0_bf, bias0, gates0, Bz, 4 * Hz, KX0);
        k_lstm_point0<<<blk(Bz * Hz, TB), TB, 0, stream>>>(gates0, c0, h0_bf, x_bf, x1_bf);
        // gates1 = [h0|h1] @ [Wih1|Whh1]^T + (bih1+bhh1) : M=32, N=4096, K=2048
        k_gemm_bf16<2><<<dim3((4 * Hz) / 64, 1), 128, 0, stream>>>(
            x1_bf, Wcat1_bf, bias1, gates1, Bz, 4 * Hz, KX1);
        k_lstm_point1<<<blk(Bz * Hz, TB), TB, 0, stream>>>(gates1, c1, x1_bf, outs_bf, t);
    }

    // ---- final projection: out = outs @ Wfc^T + bfc : M=640, N=32000, K=1024 (MT=4) ----
    k_gemm_bf16<4><<<dim3(Vz / 64, (Bz * Tz) / 64), 128, 0, stream>>>(
        outs_bf, Wfc_bf, bfc, out, Bz * Tz, Vz, Hz);
}
